// LLaMAAttention_12283606468017
// MI455X (gfx1250) — matrix operand order repaired and checked by reading, not compiled, not run on hardware
//
#include <hip/hip_runtime.h>
#include <hip/hip_bf16.h>

// ---------------------------------------------------------------------------
// LLaMA attention block for MI455X (gfx1250, wave32, WMMA bf16 16x16x32).
// Pipeline: fp32->bf16 convert (weights transposed) -> 3 WMMA GEMMs writing
// [B,H,S,HD] -> in-place RoPE -> flash attention (WMMA QK^T + PV, online
// softmax) -> WMMA GEMM with Wo -> fp32 out.
// GEMM + flash K-tiles staged with gfx1250 async global->LDS copies
// (global_load_async_to_lds_b128 / s_wait_asynccnt), GEMM double-buffered.
// ---------------------------------------------------------------------------

#define BATCH 2
#define SEQ   2048
#define DMODEL 4096
#define NHEAD 32
#define HDIM  128
#define MTOT  (BATCH * SEQ)   // 4096

typedef __bf16 bf16x16 __attribute__((ext_vector_type(16)));
typedef float  floatx8 __attribute__((ext_vector_type(8)));

union Frag16 { bf16x16 v; unsigned u[8]; };

__device__ __forceinline__ unsigned short f2bf(float f) {
  union { float f; unsigned u; } x; x.f = f;
  unsigned r = x.u + 0x7FFFu + ((x.u >> 16) & 1u);   // round-to-nearest-even
  return (unsigned short)(r >> 16);
}
__device__ __forceinline__ float bf2f(unsigned short h) {
  union { unsigned u; float f; } x; x.u = ((unsigned)h) << 16; return x.f;
}

// A-fragment (16x32 bf16) k index for vgpr v, lane-half hi
__device__ __forceinline__ int akidx(int v, int hi) {
  return ((v >> 2) << 4) + (hi << 3) + ((v & 3) << 1);
}

// Async 16B global->LDS copy (per-lane). LDS dest must be 16B aligned.
// Tracked by ASYNCcnt; completion enforced with s_wait_asynccnt.
__device__ __forceinline__ void async_copy_b128(const void* gptr, void* lds) {
  unsigned ldsoff = (unsigned)(uintptr_t)lds;   // low 32 bits = LDS byte offset
  asm volatile("global_load_async_to_lds_b128 %0, %1, off"
               :: "v"(ldsoff), "v"(gptr) : "memory");
}
__device__ __forceinline__ void wait_async0() {
  asm volatile("s_wait_asynccnt 0x0" ::: "memory");
}

// --------------------------- convert kernels -------------------------------

__global__ void cvt_bf16_kernel(const float* __restrict__ in,
                                unsigned short* __restrict__ out, int n) {
  int i = blockIdx.x * blockDim.x + threadIdx.x;
  if (i < n) out[i] = f2bf(in[i]);
}

// W is [K][N] row-major fp32; write Wt [N][K] bf16 (so B-frag K pairs contiguous)
__global__ void cvt_transpose_bf16_kernel(const float* __restrict__ in,
                                          unsigned short* __restrict__ out) {
  int i = blockIdx.x * blockDim.x + threadIdx.x;
  if (i >= DMODEL * DMODEL) return;
  int k = i >> 12;          // / 4096
  int n = i & 4095;
  out[(size_t)n * DMODEL + k] = f2bf(in[i]);
}

// ------------------------------- GEMM --------------------------------------
// C[M=4096][N=4096] = A[M][K=4096] * Bt[N][K]^T, bf16 in, fp32 accum.
// MODE 0: write bf16 into [B, H, S, HD] head layout. MODE 1: write fp32 flat.
// Double-buffered LDS, tiles staged via async global->LDS B128 copies.

#define LDSTRIDE 20   // dwords per 32-bf16 row: 80B, 16B-aligned, no bank conf.

template <int MODE>
__global__ __launch_bounds__(256)
void gemm_bf16_kernel(const unsigned short* __restrict__ A,
                      const unsigned short* __restrict__ Bt,
                      void* __restrict__ Cout) {
  __shared__ unsigned As32[2][128][LDSTRIDE];
  __shared__ unsigned Bs32[2][128][LDSTRIDE];

  const int t = threadIdx.x;
  const int w = t >> 5, lane = t & 31, ln = lane & 15, hi = lane >> 4;
  const int wm = w & 3, wn = w >> 2;              // 4 waves along M, 2 along N
  const int tm0 = blockIdx.y << 7, tn0 = blockIdx.x << 7;

  floatx8 acc[2][4];
  for (int mt = 0; mt < 2; ++mt)
    for (int nt = 0; nt < 4; ++nt)
      for (int r = 0; r < 8; ++r) acc[mt][nt][r] = 0.f;

  // per-thread staging coordinates: 2 x B128 per matrix per step
  auto issue_tiles = [&](int buf, int k0) {
    for (int i = 0; i < 2; ++i) {
      int idx = i * 256 + t;          // uint4 index, 4 uint4 per 32-elem row
      int row = idx >> 2, c4 = idx & 3;
      async_copy_b128(A + (size_t)(tm0 + row) * DMODEL + k0 + c4 * 8,
                      &As32[buf][row][c4 * 4]);
      async_copy_b128(Bt + (size_t)(tn0 + row) * DMODEL + k0 + c4 * 8,
                      &Bs32[buf][row][c4 * 4]);
    }
  };

  const int nIter = DMODEL / 32;
  issue_tiles(0, 0);

  for (int it = 0; it < nIter; ++it) {
    wait_async0();                 // this wave's pending copies landed in LDS
    __syncthreads();               // everyone's tile for buffer it&1 visible
    if (it + 1 < nIter) issue_tiles((it + 1) & 1, (it + 1) * 32);
    const int buf = it & 1;

    Frag16 af[2], bfg[4];
    for (int mt = 0; mt < 2; ++mt) {
      int row = wm * 32 + mt * 16 + ln;
      for (int v = 0; v < 8; ++v)
        af[mt].u[v] = As32[buf][row][(hi << 3) + v];
    }
    for (int nt = 0; nt < 4; ++nt) {
      int row = wn * 64 + nt * 16 + ln;
      for (int v = 0; v < 8; ++v)
        bfg[nt].u[v] = Bs32[buf][row][((hi << 4) + (v << 1)) >> 1];
    }
    for (int mt = 0; mt < 2; ++mt)
      for (int nt = 0; nt < 4; ++nt)
        acc[mt][nt] = __builtin_amdgcn_wmma_f32_16x16x32_bf16(
            false, af[mt].v, false, bfg[nt].v, (short)0, acc[mt][nt], false, false);
  }

  for (int mt = 0; mt < 2; ++mt)
    for (int nt = 0; nt < 4; ++nt)
      for (int r = 0; r < 8; ++r) {
        int m = tm0 + wm * 32 + mt * 16 + r + (hi << 3);
        int n = tn0 + wn * 64 + nt * 16 + ln;
        float val = acc[mt][nt][r];
        if (MODE == 0) {
          int bb = m >> 11, s = m & 2047, h = n >> 7, d = n & 127;
          ((unsigned short*)Cout)[(((size_t)(bb * NHEAD + h) * SEQ + s) << 7) + d] =
              f2bf(val);
        } else {
          ((float*)Cout)[(size_t)m * DMODEL + n] = val;
        }
      }
}

// ------------------------------- RoPE --------------------------------------
// In-place rotate pairs of Qh/Kh ([B,H,S,HD] bf16). total = B*H*S*(HD/2).

__global__ void rope_kernel(unsigned short* __restrict__ q,
                            const float* __restrict__ fc,
                            const float* __restrict__ fs, int total) {
  int i = blockIdx.x * blockDim.x + threadIdx.x;
  if (i >= total) return;
  int j = i & 63;
  int s = (i >> 6) & (SEQ - 1);
  int bh = i >> 17;
  size_t base = (((size_t)bh * SEQ + s) << 7) + (j << 1);
  float a = bf2f(q[base]), b = bf2f(q[base + 1]);
  float c = fc[(s << 6) + j], sn = fs[(s << 6) + j];
  q[base]     = f2bf(a * c - b * sn);
  q[base + 1] = f2bf(a * sn + b * c);
}

// --------------------------- flash attention -------------------------------
// grid: (S/64, B*H), block 128 (4 waves). Each wave owns 16 query rows.
// Online softmax; causal mask applied analytically on the diagonal tile.
// K tile staged with async global->LDS copies; V tile transposed on store.

__global__ __launch_bounds__(128)
void flash_attn_kernel(const unsigned short* __restrict__ Qh,
                       const unsigned short* __restrict__ Kh,
                       const unsigned short* __restrict__ Vh,
                       unsigned short* __restrict__ Obf) {
  __shared__ unsigned short Ks16[64][136];    // [key][dim], 272B rows (16B ali)
  __shared__ unsigned short Vt16[128][66];    // [dim][key] transposed
  __shared__ unsigned short Ps16[4][16][66];  // per-wave P scratch 16x64

  const int t = threadIdx.x, w = t >> 5, lane = t & 31, ln = lane & 15,
            hi = lane >> 4;
  const int qt = blockIdx.x, bh = blockIdx.y;
  const int q0 = qt << 6;
  const size_t seqbase = (size_t)bh * SEQ;

  // Q fragments for this wave's 16 rows (A-layout, 4 K-tiles of 32)
  Frag16 qf[4];
  {
    int row = q0 + w * 16 + ln;
    const unsigned short* qp = Qh + ((seqbase + row) << 7);
    for (int kt = 0; kt < 4; ++kt)
      for (int v = 0; v < 8; ++v)
        qf[kt].u[v] = *(const unsigned*)(qp + kt * 32 + (hi << 4) + (v << 1));
  }

  float m_i[8], l_i[8];
  floatx8 oacc[8];
  for (int r = 0; r < 8; ++r) { m_i[r] = -3e38f; l_i[r] = 0.f; }
  for (int ot = 0; ot < 8; ++ot)
    for (int r = 0; r < 8; ++r) oacc[ot][r] = 0.f;

  const float scale = 0.08838834764831845f;   // 1/sqrt(128)

  for (int kt64 = 0; kt64 <= qt; ++kt64) {
    const int kv0 = kt64 << 6;
    __syncthreads();
    // K tile 64x128 bf16 row-major via async copies (8 x B128 per thread)
    for (int i = 0; i < 8; ++i) {
      int idx = i * 128 + t;                  // uint4 index, 16 per 128-e row
      int row = idx >> 4, c = idx & 15;
      async_copy_b128(Kh + ((seqbase + kv0 + row) << 7) + c * 8,
                      &Ks16[row][c * 8]);
    }
    // V tile transposed into [dim][key] (regular loads, u16 scatter stores)
    for (int i = 0; i < 8; ++i) {
      int idx = i * 128 + t;
      int row = idx >> 4, c = idx & 15;
      uint4 vv = *(const uint4*)(Vh + ((seqbase + kv0 + row) << 7) + c * 8);
      const unsigned short* pv = (const unsigned short*)&vv;
      for (int j = 0; j < 8; ++j) Vt16[c * 8 + j][row] = pv[j];
    }
    wait_async0();
    __syncthreads();

    // S = Q * K^T  (B-fragment: n = key, k = dim; pairs contiguous in Ks16)
    floatx8 sacc[4];
    for (int nt = 0; nt < 4; ++nt)
      for (int r = 0; r < 8; ++r) sacc[nt][r] = 0.f;
    for (int nt = 0; nt < 4; ++nt) {
      int n = nt * 16 + ln;
      for (int kt = 0; kt < 4; ++kt) {
        Frag16 kb;
        for (int v = 0; v < 8; ++v)
          kb.u[v] = *(const unsigned*)&Ks16[n][kt * 32 + (hi << 4) + (v << 1)];
        sacc[nt] = __builtin_amdgcn_wmma_f32_16x16x32_bf16(
            false, qf[kt].v, false, kb.v, (short)0, sacc[nt], false, false);
      }
    }

    // scale + causal mask (only the diagonal KV tile needs it)
    const bool maskit = (kt64 == qt);
    for (int nt = 0; nt < 4; ++nt)
      for (int r = 0; r < 8; ++r) {
        float x = sacc[nt][r] * scale;
        if (maskit) {
          int mg = q0 + w * 16 + r + (hi << 3);
          int ng = kv0 + nt * 16 + ln;
          if (ng > mg) x = -3e38f;
        }
        sacc[nt][r] = x;
      }

    // online softmax; each row lives in one 16-lane half -> width-16 shuffles
    for (int r = 0; r < 8; ++r) {
      float mx = sacc[0][r];
      for (int nt = 1; nt < 4; ++nt) mx = fmaxf(mx, sacc[nt][r]);
      for (int off = 8; off >= 1; off >>= 1)
        mx = fmaxf(mx, __shfl_xor(mx, off, 16));
      float mnew = fmaxf(m_i[r], mx);
      float alpha = __expf(m_i[r] - mnew);
      m_i[r] = mnew;
      l_i[r] *= alpha;
      for (int ot = 0; ot < 8; ++ot) oacc[ot][r] *= alpha;
      float sum = 0.f;
      for (int nt = 0; nt < 4; ++nt) {
        float p = __expf(sacc[nt][r] - mnew);
        sacc[nt][r] = p;
        sum += p;
      }
      for (int off = 8; off >= 1; off >>= 1) sum += __shfl_xor(sum, off, 16);
      l_i[r] += sum;
    }

    // C-layout -> A-layout for P via per-wave LDS round trip
    for (int nt = 0; nt < 4; ++nt)
      for (int r = 0; r < 8; ++r)
        Ps16[w][r + (hi << 3)][nt * 16 + ln] = f2bf(sacc[nt][r]);
    __syncthreads();

    Frag16 pf[2];
    for (int kt2 = 0; kt2 < 2; ++kt2)
      for (int v = 0; v < 8; ++v)
        pf[kt2].u[v] = *(const unsigned*)&Ps16[w][ln][kt2 * 32 + (hi << 4) + (v << 1)];

    // O += P * V  (B-fragment: n = dim, k = key; pairs contiguous in Vt16)
    for (int ot = 0; ot < 8; ++ot) {
      int n = ot * 16 + ln;
      for (int kt2 = 0; kt2 < 2; ++kt2) {
        Frag16 vb;
        for (int v = 0; v < 8; ++v)
          vb.u[v] = *(const unsigned*)&Vt16[n][kt2 * 32 + (hi << 4) + (v << 1)];
        oacc[ot] = __builtin_amdgcn_wmma_f32_16x16x32_bf16(
            false, pf[kt2].v, false, vb.v, (short)0, oacc[ot], false, false);
      }
    }
  }

  // normalize and write attention output as [B, S, D] bf16
  const int b = bh >> 5, h = bh & 31;
  for (int r = 0; r < 8; ++r) {
    float inv = 1.0f / l_i[r];
    int sg = q0 + w * 16 + r + (hi << 3);
    size_t obase = (((size_t)(b * SEQ + sg)) << 12) + (h << 7);
    for (int ot = 0; ot < 8; ++ot)
      Obf[obase + ot * 16 + ln] = f2bf(oacc[ot][r] * inv);
  }
}

// ------------------------------ launcher -----------------------------------

extern "C" void kernel_launch(void* const* d_in, const int* in_sizes, int n_in,
                              void* d_out, int out_size, void* d_ws,
                              size_t ws_size, hipStream_t stream) {
  (void)in_sizes; (void)n_in; (void)out_size; (void)ws_size;
  const float* x  = (const float*)d_in[0];
  const float* wq = (const float*)d_in[1];
  const float* wk = (const float*)d_in[2];
  const float* wv = (const float*)d_in[3];
  const float* wo = (const float*)d_in[4];
  const float* fc = (const float*)d_in[5];
  const float* fs = (const float*)d_in[6];
  // d_in[7] (mask) and d_in[8] (start_pos==0) handled analytically in-kernel.

  char* ws = (char*)d_ws;
  const size_t MB32 = (size_t)32 << 20;
  unsigned short* Xbf = (unsigned short*)(ws);
  unsigned short* Wqt = (unsigned short*)(ws + 1 * MB32);
  unsigned short* Wkt = (unsigned short*)(ws + 2 * MB32);
  unsigned short* Wvt = (unsigned short*)(ws + 3 * MB32);
  unsigned short* Wot = (unsigned short*)(ws + 4 * MB32);
  unsigned short* Qh  = (unsigned short*)(ws + 5 * MB32);
  unsigned short* Kh  = (unsigned short*)(ws + 6 * MB32);
  unsigned short* Vh  = (unsigned short*)(ws + 7 * MB32);
  unsigned short* Obf = (unsigned short*)(ws + 8 * MB32);

  const int nElem = MTOT * DMODEL;               // 16,777,216
  cvt_bf16_kernel<<<nElem / 256, 256, 0, stream>>>(x, Xbf, nElem);
  cvt_transpose_bf16_kernel<<<nElem / 256, 256, 0, stream>>>(wq, Wqt);
  cvt_transpose_bf16_kernel<<<nElem / 256, 256, 0, stream>>>(wk, Wkt);
  cvt_transpose_bf16_kernel<<<nElem / 256, 256, 0, stream>>>(wv, Wvt);
  cvt_transpose_bf16_kernel<<<nElem / 256, 256, 0, stream>>>(wo, Wot);

  dim3 ggrid(DMODEL / 128, MTOT / 128);
  gemm_bf16_kernel<0><<<ggrid, 256, 0, stream>>>(Xbf, Wqt, Qh);
  gemm_bf16_kernel<0><<<ggrid, 256, 0, stream>>>(Xbf, Wkt, Kh);
  gemm_bf16_kernel<0><<<ggrid, 256, 0, stream>>>(Xbf, Wvt, Vh);

  const int ropeTotal = BATCH * NHEAD * SEQ * (HDIM / 2);  // 8,388,608
  rope_kernel<<<ropeTotal / 256, 256, 0, stream>>>(Qh, fc, fs, ropeTotal);
  rope_kernel<<<ropeTotal / 256, 256, 0, stream>>>(Kh, fc, fs, ropeTotal);

  flash_attn_kernel<<<dim3(SEQ / 64, BATCH * NHEAD), 128, 0, stream>>>(Qh, Kh,
                                                                       Vh, Obf);

  gemm_bf16_kernel<1><<<ggrid, 256, 0, stream>>>(Obf, Wot, (float*)d_out);
}
